// DeformableTransformer_15032385536508
// MI455X (gfx1250) — matrix operand order repaired and checked by reading, not compiled, not run on hardware
//
#include <hip/hip_runtime.h>
#include <math.h>

#define BSZ 2
#define TT  2048
#define DD  2048
#define NH  8
#define NP  4
#define NLAY 4
#define DFF 2048
#define MROWS (BSZ*TT)          // 4096

typedef __bf16 bf16_t;
typedef __attribute__((ext_vector_type(16))) __bf16 v16bf;
typedef __attribute__((ext_vector_type(8)))  __bf16 v8bf;
typedef __attribute__((ext_vector_type(8)))  float  v8f;

// ---------------------------------------------------------------------------
// Pos-embed normalization term: embn[b][t] = cumsum(mask)/(total+1e-6)*2pi
// ---------------------------------------------------------------------------
__global__ __launch_bounds__(256)
void embn_kernel(const unsigned char* __restrict__ mask, float* __restrict__ embn)
{
    int b = blockIdx.x;
    const unsigned char* m = mask + (size_t)b * TT;
    float total = 0.f;
    for (int i = 0; i < TT; ++i) total += m[i] ? 1.f : 0.f;
    float inv = 6.2831853071795864f / (total + 1e-6f);
    for (int t = threadIdx.x; t < TT; t += 256) {
        float c = 0.f;
        for (int i = 0; i <= t; ++i) c += m[i] ? 1.f : 0.f;
        embn[(size_t)b * TT + t] = c * inv;
    }
}

// ---------------------------------------------------------------------------
// x[b][t][d] = src[b][d][t]  (+ bf16 copy)
// ---------------------------------------------------------------------------
__global__ __launch_bounds__(256)
void init_x_kernel(const float* __restrict__ src, float* __restrict__ x,
                   bf16_t* __restrict__ xb)
{
    size_t idx = (size_t)blockIdx.x * 256 + threadIdx.x;
    if (idx >= (size_t)MROWS * DD) return;
    int d = idx % DD;
    int t = (idx / DD) % TT;
    int b = idx / ((size_t)TT * DD);
    float v = src[((size_t)b * DD + d) * TT + t];
    x[idx]  = v;
    xb[idx] = (bf16_t)v;
}

// ---------------------------------------------------------------------------
// qb = bf16( x + pos_sine(t,d) + level_embed[d] )
// ---------------------------------------------------------------------------
__global__ __launch_bounds__(256)
void q_kernel(const float* __restrict__ x, const float* __restrict__ embn,
              const float* __restrict__ level_embed, bf16_t* __restrict__ qb)
{
    size_t idx = (size_t)blockIdx.x * 256 + threadIdx.x;
    if (idx >= (size_t)MROWS * DD) return;
    int d   = idx % DD;
    size_t row = idx / DD;                 // b*T + t
    float e  = embn[row];
    int dp   = (d >> 1) << 1;              // 2*(d//2)
    float pr = e * __expf(-9.2103403719761836f * (float)dp / (float)DD);
    float pos = (d & 1) ? __cosf(pr) : __sinf(pr);
    qb[idx] = (bf16_t)(x[idx] + pos + level_embed[d]);
}

// ---------------------------------------------------------------------------
// Repack fp32 weight [K,N] -> bf16 B-fragment layout [N/16][K/32][32][16]
// lane l, elem e of tile(nT,kc):  B[kc*32 + (l>>4)*16 + e][nT*16 + (l&15)]
// ---------------------------------------------------------------------------
__global__ __launch_bounds__(256)
void repack_kernel(const float* __restrict__ W, bf16_t* __restrict__ Bp, int K, int N)
{
    size_t idx = (size_t)blockIdx.x * 256 + threadIdx.x;
    if (idx >= (size_t)K * N) return;
    int e    = idx & 15;
    int lane = (idx >> 4) & 31;
    int kc   = (int)((idx >> 9) % (K >> 5));
    int nT   = (int)((idx >> 9) / (K >> 5));
    int k = kc * 32 + (lane >> 4) * 16 + e;
    int n = nT * 16 + (lane & 15);
    Bp[idx] = (bf16_t)W[(size_t)k * N + n];
}

// ---------------------------------------------------------------------------
// WMMA GEMM with async LDS double-buffering.
//   C[M,N] = A[M,K](bf16 row-major) x Bp(packed bf16) + bias
// Block = 256 thr (8 waves), block tile 128x128.
// Wave (r=w>>1, c=w&1): rows r*32..+31 (2 tiles), cols c*64..+63 (4 tiles).
// Per k-chunk(32): stage A(128x32) + B(128 cols) into LDS via
// GLOBAL_LOAD_ASYNC_TO_LDS_B128 (4 async ops/wave), overlap with WMMA on the
// other buffer; sync = s_wait_asynccnt + barrier.
// ---------------------------------------------------------------------------
#define LDA_S 80                      // bytes per A row slot in LDS (64 data + pad)
#define LDB_S 48                      // bytes per B lane entry (32 data + pad)
#define A_BUF (128 * LDA_S)           // 10240 B
#define B_BUF (8 * 32 * LDB_S)        // 12288 B

__global__ __launch_bounds__(256)
void gemm_wmma_kernel(const bf16_t* __restrict__ A, const bf16_t* __restrict__ Bp,
                      const float* __restrict__ bias,
                      float* __restrict__ Cf, bf16_t* __restrict__ Cb,
                      int M, int K, int N, int relu)
{
    __shared__ __align__(16) char smA[2 * A_BUF];
    __shared__ __align__(16) char smB[2 * B_BUF];

    const int tid  = threadIdx.x;
    const int lane = tid & 31;
    const int wave = tid >> 5;
    const int lh   = lane >> 4;
    const int l15  = lane & 15;
    const int wRow = wave >> 1;           // 0..3
    const int wCol = wave & 1;            // 0..1
    const int kChunks  = K >> 5;
    const int rowBlock = blockIdx.x * 128;
    const int tileBase = blockIdx.y * 8;  // global 16-col tile index base

    // stage one 32-wide k-chunk into LDS buffer `buf` (async, 4 ops per wave)
    auto stage = [&](int kc, int buf) {
        const int k0 = kc << 5;
        {   // A: 128 rows x 64B = 512 x 16B segments
            int s = tid;
#pragma unroll
            for (int it = 0; it < 2; ++it, s += 256) {
                int row = s >> 2, ch = s & 3;
                const char* g = (const char*)(A + (size_t)(rowBlock + row) * K + k0) + ch * 16;
                unsigned laddr = (unsigned)(uintptr_t)&smA[buf * A_BUF + row * LDA_S + ch * 16];
                asm volatile("global_load_async_to_lds_b128 %0, %1, off"
                             :: "v"(laddr), "v"(g) : "memory");
            }
        }
        {   // B: 8 tiles x 32 lanes x 32B = 512 x 16B segments
            int s = tid;
#pragma unroll
            for (int it = 0; it < 2; ++it, s += 256) {
                int tile = s >> 6, ln = (s >> 1) & 31, ch = s & 1;
                const char* g = (const char*)Bp +
                    ((size_t)((tileBase + tile) * kChunks + kc) * 32 + ln) * 32 + ch * 16;
                unsigned laddr = (unsigned)(uintptr_t)&smB[buf * B_BUF + (tile * 32 + ln) * LDB_S + ch * 16];
                asm volatile("global_load_async_to_lds_b128 %0, %1, off"
                             :: "v"(laddr), "v"(g) : "memory");
            }
        }
    };

    v8f acc[2][4] = {};
    stage(0, 0);
    for (int kc = 0; kc < kChunks; ++kc) {
        const int buf = kc & 1;
        if (kc + 1 < kChunks) {
            stage(kc + 1, buf ^ 1);
            asm volatile("s_wait_asynccnt 0x4" ::: "memory");  // chunk kc landed
        } else {
            asm volatile("s_wait_asynccnt 0x0" ::: "memory");
        }
        __syncthreads();

        union { v16bf v; v8bf h[2]; } afr[2];
#pragma unroll
        for (int rt = 0; rt < 2; ++rt) {
            const char* base = &smA[buf * A_BUF + (wRow * 32 + rt * 16 + l15) * LDA_S];
            afr[rt].h[0] = *(const v8bf*)(base + lh * 32);
            afr[rt].h[1] = *(const v8bf*)(base + lh * 32 + 16);
        }
#pragma unroll
        for (int t = 0; t < 4; ++t) {
            union { v16bf v; v8bf h[2]; } bfr;
            const char* bb = &smB[buf * B_BUF + ((wCol * 4 + t) * 32 + lane) * LDB_S];
            bfr.h[0] = *(const v8bf*)(bb);
            bfr.h[1] = *(const v8bf*)(bb + 16);
            acc[0][t] = __builtin_amdgcn_wmma_f32_16x16x32_bf16(
                false, afr[0].v, false, bfr.v, (short)0, acc[0][t], false, false);
            acc[1][t] = __builtin_amdgcn_wmma_f32_16x16x32_bf16(
                false, afr[1].v, false, bfr.v, (short)0, acc[1][t], false, false);
        }
        __syncthreads();
    }

#pragma unroll
    for (int rt = 0; rt < 2; ++rt) {
#pragma unroll
        for (int t = 0; t < 4; ++t) {
            const int col = (tileBase + wCol * 4 + t) * 16 + l15;
            const float bv = bias ? bias[col] : 0.f;
#pragma unroll
            for (int r = 0; r < 8; ++r) {
                const int row = rowBlock + wRow * 32 + rt * 16 + lh * 8 + r;
                float v = acc[rt][t][r] + bv;
                if (relu) v = v > 0.f ? v : 0.f;
                if (Cf) Cf[(size_t)row * N + col] = v;
                if (Cb) Cb[(size_t)row * N + col] = (bf16_t)v;
            }
        }
    }
}

// ---------------------------------------------------------------------------
// off/attn projection: oa[row][0:32]=q@Wo+bo, oa[row][32:64]=q@Wa+ba
// ---------------------------------------------------------------------------
__global__ __launch_bounds__(256)
void offattn_kernel(const bf16_t* __restrict__ qb,
                    const float* __restrict__ Wo, const float* __restrict__ bo,
                    const float* __restrict__ Wa, const float* __restrict__ ba,
                    float* __restrict__ oa)
{
    __shared__ float qs[DD];
    __shared__ float red[256];
    const int row = blockIdx.x;
    const bf16_t* qrow = qb + (size_t)row * DD;
    for (int i = threadIdx.x; i < DD; i += 256) qs[i] = (float)qrow[i];
    __syncthreads();
    const int o    = threadIdx.x & 63;
    const int part = threadIdx.x >> 6;          // 0..3, 512 k's each
    const float* W = (o < 32) ? (Wo + o) : (Wa + (o - 32));
    float s = 0.f;
    const int k0 = part * 512;
    for (int k = k0; k < k0 + 512; ++k) s += qs[k] * W[(size_t)k * 32];
    red[threadIdx.x] = s;
    __syncthreads();
    if (part == 0) {
        float tot = red[o] + red[o + 64] + red[o + 128] + red[o + 192];
        tot += (o < 32) ? bo[o] : ba[o - 32];
        oa[(size_t)row * 64 + o] = tot;
    }
}

// ---------------------------------------------------------------------------
// softmax over P + bilinear sampling:  sampb[row][h*256+c]
//   x_sample = q + off   (valid-ratio cancels; ref = (q+0.5)/T)
// ---------------------------------------------------------------------------
__global__ __launch_bounds__(256)
void sample_kernel(const float* __restrict__ value, const float* __restrict__ oa,
                   bf16_t* __restrict__ sampb)
{
    const int row = blockIdx.x;               // b*T + q
    const int b   = row >> 11;
    const int qq  = row & (TT - 1);
    const int h   = threadIdx.x >> 5;
    const int c0  = (threadIdx.x & 31) * 8;
    const float* oar = oa + (size_t)row * 64;

    float offs[NP], lg[NP];
#pragma unroll
    for (int p = 0; p < NP; ++p) { offs[p] = oar[h * NP + p]; lg[p] = oar[32 + h * NP + p]; }
    float mx = fmaxf(fmaxf(lg[0], lg[1]), fmaxf(lg[2], lg[3]));
    float den = 0.f, wp[NP];
#pragma unroll
    for (int p = 0; p < NP; ++p) { wp[p] = __expf(lg[p] - mx); den += wp[p]; }
    float rden = 1.f / den;

    const float* vbase = value + (size_t)b * TT * DD + h * 256 + c0;
    float acc[8] = {0.f, 0.f, 0.f, 0.f, 0.f, 0.f, 0.f, 0.f};
#pragma unroll
    for (int p = 0; p < NP; ++p) {
        float xs  = (float)qq + offs[p];
        float i0f = floorf(xs);
        float w1  = xs - i0f;
        int i0 = (int)i0f, i1 = i0 + 1;
        float aw  = wp[p] * rden;
        float w0m = (1.f - w1) * ((i0 >= 0 && i0 < TT) ? 1.f : 0.f) * aw;
        float w1m = w1 * ((i1 >= 0 && i1 < TT) ? 1.f : 0.f) * aw;
        int i0c = min(max(i0, 0), TT - 1);
        int i1c = min(max(i1, 0), TT - 1);
        const float* v0 = vbase + (size_t)i0c * DD;
        const float* v1 = vbase + (size_t)i1c * DD;
#pragma unroll
        for (int c = 0; c < 8; ++c) acc[c] += w0m * v0[c] + w1m * v1[c];
    }
    bf16_t* out = sampb + (size_t)row * DD + h * 256 + c0;
#pragma unroll
    for (int c = 0; c < 8; ++c) out[c] = (bf16_t)acc[c];
}

// ---------------------------------------------------------------------------
// x = LayerNorm(x + add) * g + b  (writes fp32 + bf16)
// ---------------------------------------------------------------------------
__global__ __launch_bounds__(256)
void ln_kernel(const float* __restrict__ x, const float* __restrict__ add,
               const float* __restrict__ g, const float* __restrict__ bb,
               float* __restrict__ xo, bf16_t* __restrict__ xob)
{
    __shared__ float vals[DD];
    __shared__ float r1[256], r2[256];
    const int row = blockIdx.x;
    const float* xr = x   + (size_t)row * DD;
    const float* ar = add + (size_t)row * DD;
    float s = 0.f, s2 = 0.f;
    for (int i = threadIdx.x; i < DD; i += 256) {
        float v = xr[i] + ar[i];
        vals[i] = v; s += v; s2 += v * v;
    }
    r1[threadIdx.x] = s; r2[threadIdx.x] = s2;
    __syncthreads();
    for (int st = 128; st > 0; st >>= 1) {
        if (threadIdx.x < st) {
            r1[threadIdx.x] += r1[threadIdx.x + st];
            r2[threadIdx.x] += r2[threadIdx.x + st];
        }
        __syncthreads();
    }
    const float mu  = r1[0] * (1.f / DD);
    const float var = r2[0] * (1.f / DD) - mu * mu;
    const float rs  = rsqrtf(var + 1e-5f);
    for (int i = threadIdx.x; i < DD; i += 256) {
        float y = (vals[i] - mu) * rs * g[i] + bb[i];
        xo[(size_t)row * DD + i]  = y;
        xob[(size_t)row * DD + i] = (bf16_t)y;
    }
}

// ---------------------------------------------------------------------------
// out[0 : B*D*T] = x transposed back to (B,D,T); out[B*D*T :] = mask as float
// ---------------------------------------------------------------------------
__global__ __launch_bounds__(256)
void out_kernel(const float* __restrict__ x, const unsigned char* __restrict__ mask,
                float* __restrict__ out)
{
    size_t idx = (size_t)blockIdx.x * 256 + threadIdx.x;
    const size_t nmain = (size_t)BSZ * DD * TT;
    if (idx < nmain) {
        int t = idx % TT;
        int d = (idx / TT) % DD;
        int b = idx / ((size_t)DD * TT);
        out[idx] = x[((size_t)b * TT + t) * DD + d];
    } else if (idx < nmain + (size_t)BSZ * TT) {
        size_t m = idx - nmain;
        out[idx] = mask[m] ? 1.f : 0.f;
    }
}

// ---------------------------------------------------------------------------
extern "C" void kernel_launch(void* const* d_in, const int* in_sizes, int n_in,
                              void* d_out, int out_size, void* d_ws, size_t ws_size,
                              hipStream_t stream)
{
    (void)in_sizes; (void)n_in; (void)out_size; (void)ws_size;

    const float* src  = (const float*)d_in[0];
    const unsigned char* mask = (const unsigned char*)d_in[1];
    const float* lvl  = (const float*)d_in[2];
    const float* Wo   = (const float*)d_in[3];
    const float* bo   = (const float*)d_in[4];
    const float* Wa   = (const float*)d_in[5];
    const float* ba   = (const float*)d_in[6];
    const float* Wv   = (const float*)d_in[7];
    const float* bv   = (const float*)d_in[8];
    const float* Wout = (const float*)d_in[9];
    const float* bout = (const float*)d_in[10];
    const float* ln1g = (const float*)d_in[11];
    const float* ln1b = (const float*)d_in[12];
    const float* W1   = (const float*)d_in[13];
    const float* b1   = (const float*)d_in[14];
    const float* W2   = (const float*)d_in[15];
    const float* b2   = (const float*)d_in[16];
    const float* ln2g = (const float*)d_in[17];
    const float* ln2b = (const float*)d_in[18];

    // workspace layout
    char* ws = (char*)d_ws;
    const size_t NMD_F = (size_t)MROWS * DD * sizeof(float);   // 32 MB
    const size_t NMD_B = (size_t)MROWS * DD * sizeof(bf16_t);  // 16 MB
    float*  x     = (float*)(ws);
    bf16_t* xb    = (bf16_t*)(ws + NMD_F);
    bf16_t* qb    = (bf16_t*)(ws + NMD_F + NMD_B);             // reused as h1b
    float*  bufA  = (float*)(ws + NMD_F + 2 * NMD_B);          // value / a / h
    bf16_t* sampb = (bf16_t*)(ws + 2 * NMD_F + 2 * NMD_B);
    float*  oa    = (float*)(ws + 2 * NMD_F + 3 * NMD_B);
    bf16_t* wpack = (bf16_t*)(ws + 2 * NMD_F + 3 * NMD_B + (size_t)MROWS * 64 * sizeof(float));
    float*  embn  = (float*)((char*)wpack + (size_t)DD * DD * sizeof(bf16_t));

    const int elemBlocks = (int)(((size_t)MROWS * DD + 255) / 256);
    const int packBlocks = (int)(((size_t)DD * DD + 255) / 256);
    dim3 ggrid(MROWS / 128, DD / 128);   // (32, 16)

    embn_kernel<<<BSZ, 256, 0, stream>>>(mask, embn);
    init_x_kernel<<<elemBlocks, 256, 0, stream>>>(src, x, xb);

    for (int l = 0; l < NLAY; ++l) {
        // q = x + pos + level_embed
        q_kernel<<<elemBlocks, 256, 0, stream>>>(x, embn, lvl, qb);

        // value = x @ Wv + bv
        repack_kernel<<<packBlocks, 256, 0, stream>>>(Wv + (size_t)l * DD * DD, wpack, DD, DD);
        gemm_wmma_kernel<<<ggrid, 256, 0, stream>>>(xb, wpack, bv + (size_t)l * DD,
                                                    bufA, (bf16_t*)nullptr,
                                                    MROWS, DD, DD, 0);

        // off / attn logits
        offattn_kernel<<<MROWS, 256, 0, stream>>>(qb,
            Wo + (size_t)l * DD * 32, bo + (size_t)l * 32,
            Wa + (size_t)l * DD * 32, ba + (size_t)l * 32, oa);

        // softmax + bilinear gather
        sample_kernel<<<MROWS, 256, 0, stream>>>(bufA, oa, sampb);

        // a = sampled @ Wout + bout   (overwrites bufA)
        repack_kernel<<<packBlocks, 256, 0, stream>>>(Wout + (size_t)l * DD * DD, wpack, DD, DD);
        gemm_wmma_kernel<<<ggrid, 256, 0, stream>>>(sampb, wpack, bout + (size_t)l * DD,
                                                    bufA, (bf16_t*)nullptr,
                                                    MROWS, DD, DD, 0);

        // x = LN(x + a)
        ln_kernel<<<MROWS, 256, 0, stream>>>(x, bufA,
            ln1g + (size_t)l * DD, ln1b + (size_t)l * DD, x, xb);

        // h1 = relu(x @ W1 + b1)  (bf16 only, into qb)
        repack_kernel<<<packBlocks, 256, 0, stream>>>(W1 + (size_t)l * DD * DFF, wpack, DD, DFF);
        gemm_wmma_kernel<<<ggrid, 256, 0, stream>>>(xb, wpack, b1 + (size_t)l * DFF,
                                                    (float*)nullptr, qb,
                                                    MROWS, DD, DFF, 1);

        // h = h1 @ W2 + b2
        repack_kernel<<<packBlocks, 256, 0, stream>>>(W2 + (size_t)l * DFF * DD, wpack, DFF, DD);
        gemm_wmma_kernel<<<ggrid, 256, 0, stream>>>(qb, wpack, b2 + (size_t)l * DD,
                                                    bufA, (bf16_t*)nullptr,
                                                    MROWS, DFF, DD, 0);

        // x = LN(x + h)
        ln_kernel<<<MROWS, 256, 0, stream>>>(x, bufA,
            ln2g + (size_t)l * DD, ln2b + (size_t)l * DD, x, xb);
    }

    const size_t totalOut = (size_t)BSZ * DD * TT + (size_t)BSZ * TT;
    out_kernel<<<(int)((totalOut + 255) / 256), 256, 0, stream>>>(x, mask, (float*)d_out);
}
